// TreeLSTM_1786706395442
// MI455X (gfx1250) — matrix operand order repaired and checked by reading, not compiled, not run on hardware
//
#include <hip/hip_runtime.h>
#include <hip/hip_bf16.h>

#define H_SZ 256
#define X_SZ 256
#define C_CLS 5
#define TDEPTH 13
#define M_TREE ((1 << (TDEPTH + 1)) - 1)   // 16383 nodes per tree
#define B_TREES 8
#define N_TOT (B_TREES * M_TREE)           // 131064 nodes total

typedef __attribute__((ext_vector_type(16))) _Float16 v16h;
typedef __attribute__((ext_vector_type(8)))  _Float16 v8h;
typedef __attribute__((ext_vector_type(8)))  float    v8f;

union FragU { v16h v; v8h h[2]; };

__device__ __forceinline__ float sigmoidf_(float x) {
  return 1.0f / (1.0f + __expf(-x));
}

// Build a 16x32 f16 A-fragment for one wave lane.
// row points at the 32-half K-block for matrix row (lane&15).
// Lanes 0-15 hold K = {0..15}, lanes 16-31 hold K = {16..31}.
__device__ __forceinline__ v16h load_a_frag(const _Float16* row, int ak) {
  FragU u;
  u.h[0] = *(const v8h*)(row + ak * 16);
  u.h[1] = *(const v8h*)(row + ak * 16 + 8);
  return u.v;
}

// ---------------------------------------------------------------------------
// Generic pack: W (row-major [NoutReal][Kdim] fp32; GEMM computes X @ W.T) into
// fragment-major f16 tiles: for (kt, nt), lane l holds 16 contiguous halfs:
// element e <-> (K = kt*32 + (l>>4)*16 + e, col = nt*16 + (l&15)).
// Columns >= NoutReal are zero-padded (classifier 5 -> 16 pad).
// ---------------------------------------------------------------------------
__global__ void pack_b_kernel(const float* __restrict__ W,
                              _Float16* __restrict__ dst,
                              int Kdim, int NoutPad, int NoutReal) {
  int item = blockIdx.x * blockDim.x + threadIdx.x;
  int ntiles = NoutPad >> 4;
  int ktiles = Kdim >> 5;
  int total = ktiles * ntiles * 32;
  if (item >= total) return;
  int lane = item & 31;
  int t = item >> 5;
  int nt = t % ntiles;
  int kt = t / ntiles;
  int n = nt * 16 + (lane & 15);
  int kbase = kt * 32 + (lane >> 4) * 16;
  _Float16* out = dst + ((size_t)(kt * ntiles + nt) * 32 + lane) * 16;
#pragma unroll
  for (int e = 0; e < 16; ++e) {
    float v = (n < NoutReal) ? W[(size_t)n * Kdim + kbase + e] : 0.0f;
    out[e] = (_Float16)v;
  }
}

// ---------------------------------------------------------------------------
// Combined pack for the internal-level GEMM: one buffer where, for each of the
// 16 K-steps, all 80 N-tiles (32 from U_f_w, 48 from U_iou) are contiguous.
// This makes each wave's 10 tiles addressable with immediate offsets.
// Layout: dst[((kt*80 + nt)*32 + lane)*16 + e]
// ---------------------------------------------------------------------------
__global__ void pack_b_level_kernel(const float* __restrict__ Uf,
                                    const float* __restrict__ Uiou,
                                    _Float16* __restrict__ dst) {
  int item = blockIdx.x * blockDim.x + threadIdx.x;  // (kt, nt, lane)
  if (item >= 16 * 80 * 32) return;
  int lane = item & 31;
  int t = item >> 5;
  int nt = t % 80;
  int kt = t / 80;
  int kbase = kt * 32 + (lane >> 4) * 16;
  const float* W = (nt < 32) ? Uf : Uiou;
  int col = ((nt < 32) ? nt : (nt - 32)) * 16 + (lane & 15);
  _Float16* out = dst + ((size_t)(kt * 80 + nt) * 32 + lane) * 16;
#pragma unroll
  for (int e = 0; e < 16; ++e)
    out[e] = (_Float16)W[(size_t)col * 512 + kbase + e];
}

// ---------------------------------------------------------------------------
// Leaf kernel: 16 leaves per workgroup. iou = (emb[wid*mask] @ W_iou.T)*mask,
// then h,c = apply_node(iou, b_iou, c0).
// ---------------------------------------------------------------------------
__global__ __launch_bounds__(256) void leaf_kernel(
    const float* __restrict__ emb, const _Float16* __restrict__ Wiouh,
    const float* __restrict__ b_iou, const int* __restrict__ wordid,
    const int* __restrict__ maskv, const float* __restrict__ c0,
    _Float16* __restrict__ h_half, float* __restrict__ cbuf) {
  __shared__ __align__(16) _Float16 s_x[16 * X_SZ];     // 8 KB
  __shared__ __align__(16) _Float16 s_iou[16 * 768];    // 24 KB
  __shared__ int s_idx[16];
  __shared__ float s_mask[16];

  int tid = threadIdx.x;
  int g = blockIdx.x;                 // 4096 blocks: 8 trees * 512 tiles
  int tree = g >> 9;
  int li = g & 511;
  int heap0 = (1 << TDEPTH) - 1 + li * 16;

  if (tid < 16) {
    int idx = tree * M_TREE + heap0 + tid;
    s_idx[tid] = idx;
    s_mask[tid] = (float)maskv[idx];
  }
  __syncthreads();

  // Stage x (fp32 -> f16): 16 threads per row, 16 floats each.
  {
    int row = tid >> 4;
    int part = tid & 15;
    int idx = s_idx[row];
    int wid = wordid[idx] * maskv[idx];
    const float* src = emb + (size_t)wid * X_SZ + part * 16;
    _Float16* d = &s_x[row * X_SZ + part * 16];
#pragma unroll
    for (int e = 0; e < 16; ++e) d[e] = (_Float16)src[e];
  }
  __syncthreads();

  // GEMM: 48 N-tiles over 8 waves -> 6 per wave; K = 256 -> 8 steps.
  int lane = tid & 31;
  int wave = tid >> 5;
  int am = lane & 15;
  int ak = lane >> 4;
  v8f acc[6] = {};
  // Per-lane fragment pointer: wave's 6 tiles contiguous, +j*1KB immediates.
  const v16h* bp = (const v16h*)Wiouh + (unsigned)(wave * 6) * 32 + lane;
  for (int ks = 0; ks < 8; ++ks) {
    v16h a = load_a_frag(&s_x[am * X_SZ + ks * 32], ak);
    v16h bfr[6];
#pragma unroll
    for (int j = 0; j < 6; ++j) bfr[j] = bp[j * 32];
#pragma unroll
    for (int j = 0; j < 6; ++j)
      acc[j] = __builtin_amdgcn_wmma_f32_16x16x32_f16(
          false, a, false, bfr[j], (short)0, acc[j], false, false);
    bp += 48 * 32;  // next K-step block
  }
  __syncthreads();

  // Fragment -> LDS with mask multiply. lane: col = lane&15, rows mb..mb+7.
  int cn = lane & 15;
  int mb = (lane >> 4) * 8;
#pragma unroll
  for (int j = 0; j < 6; ++j) {
    int col0 = (wave * 6 + j) * 16;
#pragma unroll
    for (int r = 0; r < 8; ++r)
      s_iou[(mb + r) * 768 + col0 + cn] = (_Float16)(acc[j][r] * s_mask[mb + r]);
  }
  __syncthreads();

  // apply_node: 16 rows, k = tid.
  for (int m = 0; m < 16; ++m) {
    int k = tid;
    float i = (float)s_iou[m * 768 + k] + b_iou[k];
    float o = (float)s_iou[m * 768 + 256 + k] + b_iou[256 + k];
    float u = (float)s_iou[m * 768 + 512 + k] + b_iou[512 + k];
    unsigned off = (unsigned)s_idx[m] * H_SZ + k;
    float cv = sigmoidf_(i) * tanhf(u) + c0[off];
    float hv = sigmoidf_(o) * tanhf(cv);
    cbuf[off] = cv;
    h_half[off] = (_Float16)hv;
  }
}

// ---------------------------------------------------------------------------
// Internal level kernel: 16 nodes per workgroup (flat enumeration over
// 8 trees x 2^lvl nodes). h_cat = [h_l, h_r]; f = sig(h_cat@Uf.T + fb);
// iou = h_cat@Uiou.T; c = sig(i)tanh(u) + fl*c_l + fr*c_r; h = sig(o)tanh(c).
// ---------------------------------------------------------------------------
__global__ __launch_bounds__(256) void level_kernel(
    const _Float16* __restrict__ UfUiou, const float* __restrict__ U_f_b,
    const float* __restrict__ b_iou, _Float16* __restrict__ h_half,
    float* __restrict__ cbuf, int lvl) {
  __shared__ __align__(16) _Float16 s_hcat[16 * 512];   // 16 KB
  __shared__ __align__(16) _Float16 s_frag[16 * 1280];  // 40 KB: [0:512)=f, [512:1280)=iou
  __shared__ int s_idx[16], s_l[16], s_r[16], s_valid[16];

  int tid = threadIdx.x;
  int per_tree = 1 << lvl;
  int total = B_TREES << lvl;

  if (tid < 16) {
    int q = blockIdx.x * 16 + tid;
    int valid = (q < total) ? 1 : 0;
    int qq = valid ? q : 0;
    int tree = qq >> lvl;
    int heap = (per_tree - 1) + (qq & (per_tree - 1));
    int base = tree * M_TREE;
    s_idx[tid] = base + heap;
    s_l[tid] = base + 2 * heap + 1;
    s_r[tid] = base + 2 * heap + 2;
    s_valid[tid] = valid;
  }
  __syncthreads();

  // Stage h_cat: 16 rows x 512 halfs, 16 threads per row x 32 halfs.
  {
    int row = tid >> 4;
    int k0 = (tid & 15) * 32;
    const _Float16* src = (k0 < H_SZ)
        ? (h_half + (unsigned)s_l[row] * H_SZ + k0)
        : (h_half + (unsigned)s_r[row] * H_SZ + (k0 - H_SZ));
    uint4 a = *(const uint4*)(src);
    uint4 b = *(const uint4*)(src + 16);
    *(uint4*)(&s_hcat[row * 512 + k0]) = a;
    *(uint4*)(&s_hcat[row * 512 + k0 + 16]) = b;
  }
  __syncthreads();

  // GEMM: 80 N-tiles (32 f + 48 iou) over 8 waves -> 10/wave; K=512 -> 16 steps.
  int lane = tid & 31;
  int wave = tid >> 5;
  int am = lane & 15;
  int ak = lane >> 4;
  v8f acc[10] = {};
  // Combined packed buffer: per K-step, all 80 tiles contiguous; this wave's
  // 10 tiles are bp[0], bp[32], ... bp[9*32] (1 KB immediate strides).
  const v16h* bp = (const v16h*)UfUiou + (unsigned)(wave * 10) * 32 + lane;
  for (int ks = 0; ks < 16; ++ks) {
    v16h a = load_a_frag(&s_hcat[am * 512 + ks * 32], ak);
    v16h bfr[10];
#pragma unroll
    for (int j = 0; j < 10; ++j) bfr[j] = bp[j * 32];
#pragma unroll
    for (int j = 0; j < 10; ++j)
      acc[j] = __builtin_amdgcn_wmma_f32_16x16x32_f16(
          false, a, false, bfr[j], (short)0, acc[j], false, false);
    bp += 80 * 32;  // next K-step block
  }
  __syncthreads();

  int cn = lane & 15;
  int mb = (lane >> 4) * 8;
#pragma unroll
  for (int j = 0; j < 10; ++j) {
    int nt = wave * 10 + j;
    int col0 = (nt < 32) ? nt * 16 : 512 + (nt - 32) * 16;  // f | iou regions
#pragma unroll
    for (int r = 0; r < 8; ++r)
      s_frag[(mb + r) * 1280 + col0 + cn] = (_Float16)acc[j][r];
  }
  __syncthreads();

  // Elementwise gates: 16 rows, k = tid.
  for (int m = 0; m < 16; ++m) {
    if (!s_valid[m]) continue;
    int k = tid;
    const _Float16* fr0 = &s_frag[m * 1280];
    float fl = sigmoidf_((float)fr0[k] + U_f_b[k]);
    float fr = sigmoidf_((float)fr0[256 + k] + U_f_b[256 + k]);
    float cl = cbuf[(unsigned)s_l[m] * H_SZ + k];
    float cr = cbuf[(unsigned)s_r[m] * H_SZ + k];
    float cred = fl * cl + fr * cr;
    float i = (float)fr0[512 + k] + b_iou[k];
    float o = (float)fr0[768 + k] + b_iou[256 + k];
    float u = (float)fr0[1024 + k] + b_iou[512 + k];
    float cv = sigmoidf_(i) * tanhf(u) + cred;
    float hv = sigmoidf_(o) * tanhf(cv);
    unsigned off = (unsigned)s_idx[m] * H_SZ + k;
    cbuf[off] = cv;
    h_half[off] = (_Float16)hv;
  }
}

// ---------------------------------------------------------------------------
// Classifier: out = h @ lin_w.T + lin_b, lin_w padded 5 -> 16 cols.
// One wave per 16 nodes; A-frags straight from global h_half.
// ---------------------------------------------------------------------------
__global__ __launch_bounds__(256) void cls_kernel(
    const _Float16* __restrict__ h_half, const _Float16* __restrict__ linh,
    const float* __restrict__ lin_b, float* __restrict__ out) {
  int tid = threadIdx.x;
  int lane = tid & 31;
  int wave = tid >> 5;
  long tile = (long)blockIdx.x * 8 + wave;
  long base = tile * 16;
  int am = lane & 15;
  int ak = lane >> 4;
  long row = base + am;
  const _Float16* hrow = h_half + (size_t)((row < N_TOT) ? row : 0) * H_SZ;
  const v16h* bp = (const v16h*)linh + lane;
  v8f acc = {};
  for (int ks = 0; ks < 8; ++ks) {
    v16h a = load_a_frag(hrow + ks * 32, ak);
    v16h b = bp[ks * 32];
    acc = __builtin_amdgcn_wmma_f32_16x16x32_f16(
        false, a, false, b, (short)0, acc, false, false);
  }
  int cn = lane & 15;
  int mb = (lane >> 4) * 8;
  if (cn < C_CLS) {
#pragma unroll
    for (int r = 0; r < 8; ++r) {
      long rr = base + mb + r;
      if (rr < N_TOT) out[rr * C_CLS + cn] = acc[r] + lin_b[cn];
    }
  }
}

// ---------------------------------------------------------------------------
extern "C" void kernel_launch(void* const* d_in, const int* in_sizes, int n_in,
                              void* d_out, int out_size, void* d_ws, size_t ws_size,
                              hipStream_t stream) {
  const float* emb    = (const float*)d_in[0];
  const float* W_iou  = (const float*)d_in[1];
  const float* U_iou  = (const float*)d_in[2];
  const float* b_iou  = (const float*)d_in[3];
  const float* U_f_w  = (const float*)d_in[4];
  const float* U_f_b  = (const float*)d_in[5];
  const float* lin_w  = (const float*)d_in[6];
  const float* lin_b  = (const float*)d_in[7];
  const int*   wordid = (const int*)d_in[8];
  const int*   maskv  = (const int*)d_in[9];
  /* h0 = d_in[10] (never read by the recurrence) */
  const float* c0     = (const float*)d_in[11];

  char* ws = (char*)d_ws;
  size_t off = 0;
  auto take = [&](size_t bytes) { char* p = ws + off; off = (off + bytes + 255) & ~(size_t)255; return p; };

  _Float16* h_half = (_Float16*)take((size_t)N_TOT * H_SZ * 2);   // ~67 MB
  float*    cbuf   = (float*)   take((size_t)N_TOT * H_SZ * 4);   // ~134 MB
  _Float16* UfUiou = (_Float16*)take((size_t)16 * 80 * 32 * 16 * 2);  // 1.25 MB
  _Float16* Wiouh  = (_Float16*)take((size_t)256 * 768 * 2);
  _Float16* linh   = (_Float16*)take((size_t)256 * 16 * 2);

  // Weight repack (fp32 -> fragment-major f16).
  pack_b_level_kernel<<<(16 * 80 * 32 + 255) / 256, 256, 0, stream>>>(U_f_w, U_iou, UfUiou);
  pack_b_kernel<<<(8 * 48 * 32 + 255) / 256, 256, 0, stream>>>(W_iou, Wiouh, 256, 768, 768);
  pack_b_kernel<<<(8 *  1 * 32 + 255) / 256, 256, 0, stream>>>(lin_w, linh, 256, 16, C_CLS);

  // Leaves: 8 trees * 512 tiles of 16 leaves.
  leaf_kernel<<<B_TREES * 512, 256, 0, stream>>>(
      emb, Wiouh, b_iou, wordid, maskv, c0, h_half, cbuf);

  // Internal levels, deepest first (sequential dependency via stream order).
  for (int lvl = TDEPTH - 1; lvl >= 0; --lvl) {
    int total = B_TREES << lvl;
    int tiles = (total + 15) / 16;
    level_kernel<<<tiles, 256, 0, stream>>>(UfUiou, U_f_b, b_iou, h_half, cbuf, lvl);
  }

  // Classifier over all nodes.
  cls_kernel<<<(N_TOT + 16 * 8 - 1) / (16 * 8), 256, 0, stream>>>(
      h_half, linh, lin_b, (float*)d_out);
}
